// SparseIncidenceConv_32444182954026
// MI455X (gfx1250) — compile-verified, matrix operand order checked by reading, not run on hardware
//
#include <hip/hip_runtime.h>
#include <hip/hip_bf16.h>

// ---------- types ----------
typedef __attribute__((ext_vector_type(16))) __bf16   v16bf;
typedef __attribute__((ext_vector_type(8)))  float    v8f;
typedef __attribute__((ext_vector_type(4)))  float    v4f;
typedef __attribute__((ext_vector_type(4)))  unsigned v4u;

union bfu {
    v16bf v;
    v4u   q[2];
};

#define D_DIM 256
#define S_F32 260   // padded f32 row stride (floats)
#define S_BF  264   // padded bf16 row stride (elements): 528B = 4-bank shift/row

// float -> bf16, round-to-nearest-even
__device__ __forceinline__ unsigned short f2bf(float f) {
    unsigned u = __float_as_uint(f);
    unsigned r = u + 0x7FFFu + ((u >> 16) & 1u);
    return (unsigned short)(r >> 16);
}

__device__ __forceinline__ float bf2f(unsigned short h) {
    return __uint_as_float(((unsigned)h) << 16);
}

// pack 8 floats (two v4f) into 8 bf16 (one v4u, memory order)
__device__ __forceinline__ v4u pk8(v4f a, v4f b) {
    v4u r;
    r.x = (unsigned)f2bf(a.x) | ((unsigned)f2bf(a.y) << 16);
    r.y = (unsigned)f2bf(a.z) | ((unsigned)f2bf(a.w) << 16);
    r.z = (unsigned)f2bf(b.x) | ((unsigned)f2bf(b.y) << 16);
    r.w = (unsigned)f2bf(b.z) | ((unsigned)f2bf(b.w) << 16);
    return r;
}

__device__ __forceinline__ v8f wmma_bf16(v16bf a, v16bf b, v8f c) {
    return __builtin_amdgcn_wmma_f32_16x16x32_bf16(
        /*neg_a=*/false, a, /*neg_b=*/false, b,
        /*c_mod=*/(short)0, c, /*reuse_a=*/false, /*reuse_b=*/false);
}

// B fragment: lane holds column n, 16 K-values of k-tile `kt`.
// WT is N-major bf16 [256][256]: WT[n][k] = W[k][n]. Two contiguous 16B loads.
__device__ __forceinline__ v16bf loadB(const unsigned short* __restrict__ WT,
                                       int n, int kt, int half) {
    bfu B;
    const unsigned short* p = WT + n * D_DIM + kt * 32 + half * 8;
    B.q[0] = *(const v4u*)(p);
    B.q[1] = *(const v4u*)(p + 16);
    return B.v;
}

// A fragment from bf16 LDS tile (row-major, stride S_BF): two 16B ds loads.
__device__ __forceinline__ v16bf loadA_lds(const unsigned short* __restrict__ T,
                                           int lr, int kt, int half) {
    bfu A;
    const unsigned short* p = T + lr * S_BF + kt * 32 + half * 8;
    A.q[0] = *(const v4u*)(p);
    A.q[1] = *(const v4u*)(p + 16);
    return A.v;
}

// ---------- kernel 1: zero accumulator (d_out) + counts ----------
__global__ void k_zero(float* __restrict__ acc, float* __restrict__ cnt,
                       long long n_acc, int n_cnt) {
    long long t = (long long)blockIdx.x * blockDim.x + threadIdx.x;
    if (t < n_acc) acc[t] = 0.0f;
    if (t < n_cnt) cnt[t] = 0.0f;
}

// ---------- kernel 2: edge scatter (segment sum + counts) ----------
__global__ void k_scatter(const float* __restrict__ Xsrc,
                          const int* __restrict__ inc_tgt,
                          const int* __restrict__ inc_src,
                          float* __restrict__ s, float* __restrict__ cnt, int E) {
    long long t = (long long)blockIdx.x * blockDim.x + threadIdx.x;
    long long total = (long long)E * 32;
    if (t >= total) return;
    int e   = (int)(t >> 5);
    int seg = (int)(t & 31);
    int sr = inc_src[e];
    int tr = inc_tgt[e];
    const float* src = Xsrc + (long long)sr * D_DIM + seg * 8;
    float*       dst = s    + (long long)tr * D_DIM + seg * 8;
    v4f a = *(const v4f*)(src);
    v4f b = *(const v4f*)(src + 4);
#define ATOM(p, v) __hip_atomic_fetch_add((p), (v), __ATOMIC_RELAXED, __HIP_MEMORY_SCOPE_AGENT)
    ATOM(dst + 0, a.x); ATOM(dst + 1, a.y); ATOM(dst + 2, a.z); ATOM(dst + 3, a.w);
    ATOM(dst + 4, b.x); ATOM(dst + 5, b.y); ATOM(dst + 6, b.z); ATOM(dst + 7, b.w);
    if (seg == 0) ATOM(cnt + tr, 1.0f);
#undef ATOM
}

// ---------- kernel 3: transpose weights to N-major bf16 ----------
__global__ void k_repack(const float* __restrict__ Wt, const float* __restrict__ Ws,
                         const float* __restrict__ gW,
                         unsigned short* __restrict__ WtT, unsigned short* __restrict__ WsT,
                         unsigned short* __restrict__ GtT, unsigned short* __restrict__ GsT) {
    int t = blockIdx.x * blockDim.x + threadIdx.x;   // 0 .. 4*65536-1
    if (t >= 4 * D_DIM * D_DIM) return;
    int mat = t >> 16;
    int idx = t & 0xFFFF;
    int n = idx >> 8;
    int k = idx & 255;
    float v;
    unsigned short* dst;
    if (mat == 0)      { v = Wt[k * D_DIM + n];            dst = WtT; }
    else if (mat == 1) { v = Ws[k * D_DIM + n];            dst = WsT; }
    else if (mat == 2) { v = gW[k * D_DIM + n];            dst = GtT; }
    else               { v = gW[(D_DIM + k) * D_DIM + n];  dst = GsT; }
    dst[n * D_DIM + k] = f2bf(v);
}

// ---------- kernel 4: fused GEMMs + gate + mix + layernorm ----------
// 4 waves / block, one 16-row tile per wave, no inter-wave sharing (no barriers).
__global__ __launch_bounds__(128)
void k_fused(const float* __restrict__ Xtgt,
             float* __restrict__ agg_out,          // agg accumulator, becomes output
             const float* __restrict__ cnt,
             const unsigned short* __restrict__ WtT,
             const unsigned short* __restrict__ WsT,
             const unsigned short* __restrict__ GtT,
             const unsigned short* __restrict__ GsT,
             const float* __restrict__ gate_b,
             const float* __restrict__ ln_g,
             const float* __restrict__ ln_b,
             int Ntgt) {
    // per wave: ht/hs tiles as bf16 (A-ready, row-major), mixed tile as f32
    __shared__ unsigned short hbuf[4 * 2 * 16 * S_BF];   //  67.6 KB
    __shared__ float          mbuf[4 * 16 * S_F32];      //  66.6 KB

    const int wid  = threadIdx.x >> 5;
    const int lane = threadIdx.x & 31;
    const int half = lane >> 4;      // 0: lanes 0-15, 1: lanes 16-31
    const int lr   = lane & 15;

    const int tile = blockIdx.x * 4 + wid;
    const int m0 = tile * 16;
    if (m0 >= Ntgt) return;          // wave-uniform

    unsigned short* htb = hbuf + wid * (2 * 16 * S_BF);
    unsigned short* hsb = htb + 16 * S_BF;
    float*          mx  = mbuf + wid * (16 * S_F32);

    // ---------------- pass 1: A fragments for X_tgt and agg ----------------
    const int row = m0 + lr;
    const float invc = 1.0f / (cnt[row] + 1e-8f);
    const float* xt = Xtgt    + (long long)row * D_DIM;
    const float* xa = agg_out + (long long)row * D_DIM;

    v16bf At[8], As[8];
#pragma unroll
    for (int kt = 0; kt < 8; ++kt) {
        int k0 = kt * 32 + half * 8;
        {
            v4f p = *(const v4f*)(xt + k0);
            v4f q = *(const v4f*)(xt + k0 + 4);
            v4f r = *(const v4f*)(xt + k0 + 16);
            v4f w = *(const v4f*)(xt + k0 + 20);
            bfu A; A.q[0] = pk8(p, q); A.q[1] = pk8(r, w);
            At[kt] = A.v;
        }
        {
            v4f p = *(const v4f*)(xa + k0);
            v4f q = *(const v4f*)(xa + k0 + 4);
            v4f r = *(const v4f*)(xa + k0 + 16);
            v4f w = *(const v4f*)(xa + k0 + 20);
            p *= invc; q *= invc; r *= invc; w *= invc;
            bfu A; A.q[0] = pk8(p, q); A.q[1] = pk8(r, w);
            As[kt] = A.v;
        }
    }

    for (int nt = 0; nt < 16; ++nt) {
        const int n0 = nt * 16;
        v8f ct = {0, 0, 0, 0, 0, 0, 0, 0};
        v8f cs = {0, 0, 0, 0, 0, 0, 0, 0};
#pragma unroll
        for (int kt = 0; kt < 8; ++kt) {
            ct = wmma_bf16(At[kt], loadB(WtT, n0 + lr, kt, half), ct);
            cs = wmma_bf16(As[kt], loadB(WsT, n0 + lr, kt, half), cs);
        }
        // C-fragment -> bf16 LDS tiles (A-ready layout for pass 2)
#pragma unroll
        for (int v = 0; v < 8; ++v) {
            int m = v + 8 * half;
            htb[m * S_BF + n0 + lr] = f2bf(ct[v]);
            hsb[m * S_BF + n0 + lr] = f2bf(cs[v]);
        }
    }

    // ---------------- pass 2: gate GEMM (bias-seeded C), sigmoid, mix ----------------
    for (int nt = 0; nt < 16; ++nt) {
        const int n0 = nt * 16;
        const int col = n0 + lr;
        const float bias = gate_b[col];
        v8f acc = {bias, bias, bias, bias, bias, bias, bias, bias};
#pragma unroll
        for (int kt = 0; kt < 8; ++kt) {
            acc = wmma_bf16(loadA_lds(htb, lr, kt, half),
                            loadB(GtT, col, kt, half), acc);
        }
#pragma unroll
        for (int kt = 0; kt < 8; ++kt) {
            acc = wmma_bf16(loadA_lds(hsb, lr, kt, half),
                            loadB(GsT, col, kt, half), acc);
        }
#pragma unroll
        for (int v = 0; v < 8; ++v) {
            int m = v + 8 * half;
            float g = 1.0f / (1.0f + __expf(-acc[v]));
            float a = bf2f(htb[m * S_BF + col]);
            float b = bf2f(hsb[m * S_BF + col]);
            mx[m * S_F32 + col] = g * a + (1.0f - g) * b;
        }
    }

    // ---------------- pass 3: layernorm (2 lanes per row) ----------------
    {
        const float* mrow = mx + lr * S_F32 + half * 128;
        float sum = 0.0f, sq = 0.0f;
#pragma unroll 8
        for (int i = 0; i < 128; ++i) {
            float x = mrow[i];
            sum += x;
            sq += x * x;
        }
        sum += __shfl_xor(sum, 16, 32);
        sq  += __shfl_xor(sq, 16, 32);
        const float mu   = sum * (1.0f / 256.0f);
        const float var  = sq * (1.0f / 256.0f) - mu * mu;
        const float rstd = rsqrtf(var + 1e-5f);
        float* orow = agg_out + (long long)(m0 + lr) * D_DIM;
#pragma unroll 4
        for (int i = 0; i < 128; ++i) {
            int c = half * 128 + i;
            float x = mrow[i];
            orow[c] = (x - mu) * rstd * ln_g[c] + ln_b[c];
        }
    }
}

// ---------- host launcher ----------
extern "C" void kernel_launch(void* const* d_in, const int* in_sizes, int n_in,
                              void* d_out, int out_size, void* d_ws, size_t ws_size,
                              hipStream_t stream) {
    const float* Xt   = (const float*)d_in[0];
    const float* Xs   = (const float*)d_in[1];
    const int*   itgt = (const int*)d_in[2];
    const int*   isrc = (const int*)d_in[3];
    // d_in[4] = num_tgt scalar (unused; derive from sizes)
    const float* Wt   = (const float*)d_in[5];
    const float* Ws   = (const float*)d_in[6];
    const float* gW   = (const float*)d_in[7];
    const float* gb   = (const float*)d_in[8];
    const float* lng  = (const float*)d_in[9];
    const float* lnb  = (const float*)d_in[10];

    const int Ntgt = in_sizes[0] / D_DIM;
    const int E    = in_sizes[2];

    float* out = (float*)d_out;                 // doubles as agg accumulator
    float* cnt = (float*)d_ws;
    size_t off = ((size_t)Ntgt * 4 + 255) & ~(size_t)255;
    unsigned short* WtT = (unsigned short*)((char*)d_ws + off);
    unsigned short* WsT = WtT + D_DIM * D_DIM;
    unsigned short* GtT = WsT + D_DIM * D_DIM;
    unsigned short* GsT = GtT + D_DIM * D_DIM;

    // 1) zero accumulator + counts
    long long n_acc = (long long)Ntgt * D_DIM;
    int zb = (int)((n_acc + 255) / 256);
    k_zero<<<zb, 256, 0, stream>>>(out, cnt, n_acc, Ntgt);

    // 2) scatter
    long long sthreads = (long long)E * 32;
    int sb = (int)((sthreads + 255) / 256);
    k_scatter<<<sb, 256, 0, stream>>>(Xs, itgt, isrc, out, cnt, E);

    // 3) repack weights to N-major bf16
    k_repack<<<(4 * D_DIM * D_DIM + 255) / 256, 256, 0, stream>>>(Wt, Ws, gW, WtT, WsT, GtT, GsT);

    // 4) fused GEMMs + gate + layernorm (one wave per 16-row tile, 4 waves/block)
    int tiles = (Ntgt + 15) / 16;
    int fb = (tiles + 3) / 4;
    k_fused<<<fb, 128, 0, stream>>>(Xt, out, cnt, WtT, WsT, GtT, GsT, gb, lng, lnb, Ntgt);
}
